// NonLocalBlock_23450521436742
// MI455X (gfx1250) — matrix operand order repaired and checked by reading, not compiled, not run on hardware
//
#include <hip/hip_runtime.h>

typedef _Float16 half_t;
typedef __attribute__((ext_vector_type(16))) _Float16 v16h;
typedef __attribute__((ext_vector_type(8)))  float    v8f;

#define BMT 64    // block tile rows
#define BNT 128   // block tile cols
#define BK  32    // K step (one 16x16x32 WMMA K-slab)
#define BKP 40    // padded LDS row length in halfs (80 B, multiple of 16 B)

#define MODE_F32         0
#define MODE_F16_BIAS    1   // f16 out, + bias[row]
#define MODE_F16_BIASCOL 2   // f16 out, + bias[col]   (transposed-output projection)
#define MODE_F16         3   // f16 out, no bias
#define MODE_FUSED       4   // fp32 out = relu(xres + acc + bias[row])

// ---------------------------------------------------------------------------
// LDS-tiled WMMA GEMM:  D[M x Ncols] = A[M x K] * B[K x Ncols]
//   A : f16 row-major, lda in halfs.
//   B : supplied K-CONTIGUOUS, i.e. memory holds Bt[Ncols x K] row-major
//       (B[k,n] = Bt[n*ldb + k]) -> all global->LDS traffic is b128 copies.
// Block = 128 threads / 4 waves; each wave computes 32x64 (2x4 WMMA tiles):
// 8 v_wmma per K-step vs 12 ds_load_b128 fragment loads.
// Next K-step's global tile is prefetched into NAMED uint4 scalars (arrays
// here fail SROA and get demoted to scratch/LDS allocas -- observed in asm),
// so the prefetch stays in VGPRs and overlaps the WMMA body.
// M mult of 64, Ncols mult of 128, K mult of 32 (true for all uses here).
// ---------------------------------------------------------------------------
template<int MODE>
__global__ __launch_bounds__(128, 1)
void nl_gemm_wmma(const half_t* __restrict__ A, int lda,
                  const half_t* __restrict__ Bt, int ldb,
                  void* __restrict__ Dv, int ldd, int K,
                  const float* __restrict__ bias,
                  const float* __restrict__ xres)
{
    __shared__ __align__(16) half_t lA[BMT * BKP];
    __shared__ __align__(16) half_t lB[BNT * BKP];   // K-contiguous per output col

    const int tid  = threadIdx.x;
    const int lane = tid & 31;
    const int wave = tid >> 5;
    const int wm   = (wave >> 1) * 32;   // wave row offset in block tile
    const int wn   = (wave & 1) * 64;    // wave col offset in block tile

    const int m0 = blockIdx.y * BMT;
    const int n0 = blockIdx.x * BNT;

    const int lr = lane & 15;
    const int lh = lane >> 4;

    // global->LDS coordinates
    const int arow  = tid >> 1;          // 0..63
    const int apart = (tid & 1) * 16;    // 0 or 16 halfs

    const half_t* gA = A  + (size_t)(m0 + arow) * lda + apart;  // +k
    const half_t* gB = Bt + (size_t)(n0 + tid)  * ldb;          // +k (32 halfs/row)

    // Prefetch staging: named scalars (NOT arrays) so they stay in VGPRs.
    uint4 ra0, ra1, rb0, rb1, rb2, rb3;
    { // prefetch K-step 0
        const uint4* s = (const uint4*)gA; ra0 = s[0]; ra1 = s[1];
        const uint4* t = (const uint4*)gB;
        rb0 = t[0]; rb1 = t[1]; rb2 = t[2]; rb3 = t[3];
    }

    v8f acc[2][4] = {};
    union V16 { v16h v; struct { uint4 lo, hi; } q; };

    const int nk = K / BK;
    for (int kt = 0; kt < nk; ++kt) {
        __syncthreads();   // previous iteration's fragment reads done
        { uint4* d = (uint4*)(&lA[arow * BKP + apart]); d[0] = ra0; d[1] = ra1; }
        { uint4* d = (uint4*)(&lB[tid * BKP]);
          d[0] = rb0; d[1] = rb1; d[2] = rb2; d[3] = rb3; }
        if (kt + 1 < nk) { // prefetch next K-step; overlaps with WMMA below
            const uint4* s = (const uint4*)(gA + (size_t)(kt + 1) * BK);
            ra0 = s[0]; ra1 = s[1];
            const uint4* t = (const uint4*)(gB + (size_t)(kt + 1) * BK);
            rb0 = t[0]; rb1 = t[1]; rb2 = t[2]; rb3 = t[3];
        }
        __syncthreads();

        // A fragments (ISA 16-bit A 16x32): lane<16 -> K 0-7 / 16-23,
        //                                   lane>=16 -> K 8-15 / 24-31.
        V16 af[2], bf[4];
        #pragma unroll
        for (int mi = 0; mi < 2; ++mi) {
            const half_t* base = &lA[(wm + mi * 16 + lr) * BKP];
            const int kb = lh * 8;
            af[mi].q.lo = *(const uint4*)(base + kb);
            af[mi].q.hi = *(const uint4*)(base + kb + 16);
        }
        // B fragments: lane<16 -> col lr, K 0-7 / 16-23; lane>=16 -> K 8-15 / 24-31.
        #pragma unroll
        for (int ni = 0; ni < 4; ++ni) {
            const half_t* base = &lB[(wn + ni * 16 + lr) * BKP];
            const int kb = lh * 8;
            bf[ni].q.lo = *(const uint4*)(base + kb);
            bf[ni].q.hi = *(const uint4*)(base + kb + 16);
        }
        #pragma unroll
        for (int mi = 0; mi < 2; ++mi)
            #pragma unroll
            for (int ni = 0; ni < 4; ++ni)
                acc[mi][ni] = __builtin_amdgcn_wmma_f32_16x16x32_f16(
                    false, af[mi].v, false, bf[ni].v,
                    (short)0, acc[mi][ni], false, false);
    }

    // C/D layout: VGPR e -> M = e (+8 for lanes 16-31), N = lane&15
    #pragma unroll
    for (int mi = 0; mi < 2; ++mi) {
        #pragma unroll
        for (int ni = 0; ni < 4; ++ni) {
            const int col  = n0 + wn + ni * 16 + lr;
            const int rowb = m0 + wm + mi * 16 + lh * 8;
            #pragma unroll
            for (int e = 0; e < 8; ++e) {
                const int row = rowb + e;
                const float v = acc[mi][ni][e];
                const size_t idx = (size_t)row * ldd + col;
                if (MODE == MODE_F32) {
                    ((float*)Dv)[idx] = v;
                } else if (MODE == MODE_F16_BIAS) {
                    ((half_t*)Dv)[idx] = (half_t)(v + bias[row]);
                } else if (MODE == MODE_F16_BIASCOL) {
                    ((half_t*)Dv)[idx] = (half_t)(v + bias[col]);
                } else if (MODE == MODE_F16) {
                    ((half_t*)Dv)[idx] = (half_t)v;
                } else { // MODE_FUSED
                    float r = v + bias[row] + xres[idx];
                    ((float*)Dv)[idx] = r > 0.f ? r : 0.f;
                }
            }
        }
    }
}

// ---------------------------------------------------------------------------
// fp32 -> f16 conversion (weights)
// ---------------------------------------------------------------------------
__global__ __launch_bounds__(256)
void nl_cvt_f32_f16(const float* __restrict__ s, half_t* __restrict__ d, int n)
{
    int i = blockIdx.x * 256 + threadIdx.x;
    if (i < n) d[i] = (half_t)s[i];
}

// ---------------------------------------------------------------------------
// Fused fp32->f16 transpose: src fp32 [R x Cc] row-major -> dst f16 [Cc x R].
// 64x64 LDS tile; both global sides fully coalesced.
// ---------------------------------------------------------------------------
__global__ __launch_bounds__(256)
void nl_cvtT_f32_f16(const float* __restrict__ src, half_t* __restrict__ dst,
                     int R, int Cc)
{
    __shared__ __align__(16) half_t t[64 * 72];
    const int r0 = blockIdx.y * 64, c0 = blockIdx.x * 64;
    const int tr = threadIdx.x >> 2, tc = (threadIdx.x & 3) * 16;
    const float* s = src + (size_t)(r0 + tr) * Cc + c0 + tc;
    half_t* lp = &t[tr * 72 + tc];
    #pragma unroll
    for (int j = 0; j < 16; ++j) lp[j] = (half_t)s[j];
    __syncthreads();
    union { uint4 u[2]; half_t h[16]; } w;
    #pragma unroll
    for (int j = 0; j < 16; ++j) w.h[j] = t[(tc + j) * 72 + tr];
    uint4* d = (uint4*)(dst + (size_t)(c0 + tr) * R + r0 + tc);
    d[0] = w.u[0]; d[1] = w.u[1];
}

// ---------------------------------------------------------------------------
// f16 transpose: src [R x Cc] row-major -> dst [Cc x R] row-major.
// ---------------------------------------------------------------------------
__global__ __launch_bounds__(256)
void nl_transpose_f16(const half_t* __restrict__ src, half_t* __restrict__ dst,
                      int R, int Cc)
{
    __shared__ __align__(16) half_t t[64 * 72];
    const int r0 = blockIdx.y * 64, c0 = blockIdx.x * 64;
    const int tr = threadIdx.x >> 2, tc = (threadIdx.x & 3) * 16;
    { union { uint4 u[2]; half_t h[16]; } v;
      const uint4* s = (const uint4*)(src + (size_t)(r0 + tr) * Cc + c0 + tc);
      v.u[0] = s[0]; v.u[1] = s[1];
      uint4* lp = (uint4*)(&t[tr * 72 + tc]);
      lp[0] = v.u[0]; lp[1] = v.u[1]; }
    __syncthreads();
    union { uint4 u[2]; half_t h[16]; } w;
    #pragma unroll
    for (int j = 0; j < 16; ++j) w.h[j] = t[(tc + j) * 72 + tr];
    uint4* d = (uint4*)(dst + (size_t)(c0 + tr) * R + r0 + tc);
    d[0] = w.u[0]; d[1] = w.u[1];
}

// ---------------------------------------------------------------------------
// Row softmax: one block per row of S[N x N]; exp values held in registers.
// ---------------------------------------------------------------------------
__global__ __launch_bounds__(256)
void nl_softmax_row(const float* __restrict__ S, half_t* __restrict__ O, int N)
{
    __shared__ float red[256];
    const int row = blockIdx.x;
    const int t   = threadIdx.x;
    const float* sr = S + (size_t)row * N;
    float v[16];
    float mx = -3.4e38f;
    #pragma unroll
    for (int j = 0; j < 16; ++j) { v[j] = sr[t + j * 256]; mx = fmaxf(mx, v[j]); }
    red[t] = mx; __syncthreads();
    for (int s = 128; s > 0; s >>= 1) {
        if (t < s) red[t] = fmaxf(red[t], red[t + s]);
        __syncthreads();
    }
    mx = red[0]; __syncthreads();
    float sum = 0.f;
    #pragma unroll
    for (int j = 0; j < 16; ++j) { v[j] = __expf(v[j] - mx); sum += v[j]; }
    red[t] = sum; __syncthreads();
    for (int s = 128; s > 0; s >>= 1) {
        if (t < s) red[t] += red[t + s];
        __syncthreads();
    }
    const float inv = 1.f / red[0];
    half_t* orow = O + (size_t)row * N;
    #pragma unroll
    for (int j = 0; j < 16; ++j) orow[t + j * 256] = (half_t)(v[j] * inv);
}

// ---------------------------------------------------------------------------
// Column softmax over S[N x N] (reduce over rows), 4 passes, S L2-resident.
// ---------------------------------------------------------------------------
__global__ __launch_bounds__(256)
void nl_colmax_part(const float* __restrict__ S, float* __restrict__ pmax,
                    int N, int rowsPer)
{
    const int m  = blockIdx.x * 256 + threadIdx.x;
    const int r0 = blockIdx.y * rowsPer;
    float mx = -3.4e38f;
    for (int i = 0; i < rowsPer; ++i) mx = fmaxf(mx, S[(size_t)(r0 + i) * N + m]);
    pmax[(size_t)blockIdx.y * N + m] = mx;
}

__global__ __launch_bounds__(256)
void nl_colmax_fin(const float* __restrict__ pmax, float* __restrict__ cmax,
                   float* __restrict__ csum, int N, int parts)
{
    const int m = blockIdx.x * 256 + threadIdx.x;
    float mx = -3.4e38f;
    for (int p = 0; p < parts; ++p) mx = fmaxf(mx, pmax[(size_t)p * N + m]);
    cmax[m] = mx;
    csum[m] = 0.f;
}

__global__ __launch_bounds__(256)
void nl_colsum_part(const float* __restrict__ S, const float* __restrict__ cmax,
                    float* __restrict__ csum, int N, int rowsPer)
{
    const int m  = blockIdx.x * 256 + threadIdx.x;
    const int r0 = blockIdx.y * rowsPer;
    const float mx = cmax[m];
    float s = 0.f;
    for (int i = 0; i < rowsPer; ++i) s += __expf(S[(size_t)(r0 + i) * N + m] - mx);
    atomicAdd(&csum[m], s);
}

__global__ __launch_bounds__(256)
void nl_colwrite(const float* __restrict__ S, const float* __restrict__ cmax,
                 const float* __restrict__ csum, half_t* __restrict__ O,
                 int N, int rowsPer)
{
    const int m  = blockIdx.x * 256 + threadIdx.x;
    const int r0 = blockIdx.y * rowsPer;
    const float mx  = cmax[m];
    const float inv = 1.f / csum[m];
    for (int i = 0; i < rowsPer; ++i) {
        const size_t idx = (size_t)(r0 + i) * N + m;
        O[idx] = (half_t)(__expf(S[idx] - mx) * inv);
    }
}

// ---------------------------------------------------------------------------
extern "C" void kernel_launch(void* const* d_in, const int* in_sizes, int n_in,
                              void* d_out, int out_size, void* d_ws, size_t ws_size,
                              hipStream_t stream)
{
    const int C = 256, N = 4096, Bsz = 4;

    const float* x      = (const float*)d_in[0];
    const float* w_teta = (const float*)d_in[1];
    const float* b_teta = (const float*)d_in[2];
    const float* w_fi   = (const float*)d_in[3];
    const float* b_fi   = (const float*)d_in[4];
    const float* w_gi   = (const float*)d_in[5];
    const float* b_gi   = (const float*)d_in[6];
    const float* w_o1   = (const float*)d_in[7];
    const float* b_o1   = (const float*)d_in[8];
    const float* w_o2   = (const float*)d_in[9];
    const float* b_o2   = (const float*)d_in[10];
    float* out = (float*)d_out;

    size_t off = 0;
    auto alloc = [&](size_t bytes) -> void* {
        void* p = (char*)d_ws + off;
        off += (bytes + 255) & ~(size_t)255;
        return p;
    };
    const int wn = C * C;
    half_t* xhT   = (half_t*)alloc((size_t)N * C * 2);   // x^T in f16, per batch
    half_t* wh    = (half_t*)alloc((size_t)5 * wn * 2);  // teta, fi, gi, o1, o2
    half_t* P1    = (half_t*)alloc((size_t)C * N * 2);   // natural [C,N]
    half_t* P2t   = (half_t*)alloc((size_t)N * C * 2);   // transposed: P2t[n,ch]
    half_t* P3    = (half_t*)alloc((size_t)C * N * 2);   // natural [C,N]
    half_t* x3vT  = (half_t*)alloc((size_t)C * N * 2);   // transpose of viewNC(P3)
    float*  S     = (float*)alloc((size_t)N * N * 4);
    half_t* Srow  = (half_t*)alloc((size_t)N * N * 2);
    half_t* Scol  = (half_t*)alloc((size_t)N * N * 2);
    half_t* y1    = (half_t*)alloc((size_t)N * C * 2);
    half_t* y2    = (half_t*)alloc((size_t)N * C * 2);
    float*  pmax  = (float*)alloc((size_t)32 * N * 4);
    float*  cmax  = (float*)alloc((size_t)N * 4);
    float*  csum  = (float*)alloc((size_t)N * 4);
    (void)ws_size; (void)in_sizes; (void)n_in; (void)out_size;

    // Convert the 5 weight matrices to f16 once.
    const int cb = (wn + 255) / 256;
    nl_cvt_f32_f16<<<cb, 256, 0, stream>>>(w_teta, wh + 0 * (size_t)wn, wn);
    nl_cvt_f32_f16<<<cb, 256, 0, stream>>>(w_fi,   wh + 1 * (size_t)wn, wn);
    nl_cvt_f32_f16<<<cb, 256, 0, stream>>>(w_gi,   wh + 2 * (size_t)wn, wn);
    nl_cvt_f32_f16<<<cb, 256, 0, stream>>>(w_o1,   wh + 3 * (size_t)wn, wn);
    nl_cvt_f32_f16<<<cb, 256, 0, stream>>>(w_o2,   wh + 4 * (size_t)wn, wn);

    const dim3 gProjN(N / BNT, C / BMT);   // P1, P3:  M=C,  Ncols=N, K=C
    const dim3 gProjT(C / BNT, N / BMT);   // P2t:     M=N,  Ncols=C, K=C
    const dim3 gS(N / BNT, N / BMT);       // scores:  M=N,  Ncols=N, K=C
    const dim3 gY(C / BNT, N / BMT);       // attn*V:  M=N,  Ncols=C, K=N
    const dim3 gO(N / BNT, C / BMT);       // out:     M=C,  Ncols=N, K=C

    for (int b = 0; b < Bsz; ++b) {
        const float* xb = x + (size_t)b * C * N;

        // xhT[n,c] = (f16) x[c,n]
        nl_cvtT_f32_f16<<<dim3(N / 64, C / 64), 256, 0, stream>>>(xb, xhT, C, N);

        // P1[ch,n] = W_teta * x + b   (Bt = xhT, ldb=C)
        nl_gemm_wmma<MODE_F16_BIAS><<<gProjN, 128, 0, stream>>>(
            wh + 0 * (size_t)wn, C, xhT, C, P1, N, C, b_teta, nullptr);
        // P3[ch,n] = W_gi * x + b
        nl_gemm_wmma<MODE_F16_BIAS><<<gProjN, 128, 0, stream>>>(
            wh + 2 * (size_t)wn, C, xhT, C, P3, N, C, b_gi, nullptr);
        // P2t[n,ch] = (W_fi * x + b)^T = xhT * W_fi^T   (Bt = W_fi natural, ldb=C)
        nl_gemm_wmma<MODE_F16_BIASCOL><<<gProjT, 128, 0, stream>>>(
            xhT, C, wh + 1 * (size_t)wn, C, P2t, C, C, b_fi, nullptr);

        // scores[i,m] = viewNC(P1)[i,:] . P2[:,m]   (A lda=C view; Bt=P2t ldb=C)
        nl_gemm_wmma<MODE_F32><<<gS, 128, 0, stream>>>(
            P1, C, P2t, C, S, N, C, nullptr, nullptr);

        // softmaxes
        nl_softmax_row<<<N, 256, 0, stream>>>(S, Srow, N);
        nl_colmax_part<<<dim3(N / 256, 32), 256, 0, stream>>>(S, pmax, N, 128);
        nl_colmax_fin<<<N / 256, 256, 0, stream>>>(pmax, cmax, csum, N, 32);
        nl_colsum_part<<<dim3(N / 256, 32), 256, 0, stream>>>(S, cmax, csum, N, 128);
        nl_colwrite<<<dim3(N / 256, 32), 256, 0, stream>>>(S, cmax, csum, Scol, N, 128);

        // x3vT[c,m] = viewNC(P3)[m,c]: transpose flat P3 read as [N x C]
        nl_transpose_f16<<<dim3(C / 64, N / 64), 256, 0, stream>>>(P3, x3vT, N, C);

        // y{1,2}[i,c] = S{row,col} . viewNC(P3)   (Bt = x3vT, ldb=N)
        nl_gemm_wmma<MODE_F16><<<gY, 128, 0, stream>>>(
            Srow, N, x3vT, N, y1, C, N, nullptr, nullptr);
        nl_gemm_wmma<MODE_F16><<<gY, 128, 0, stream>>>(
            Scol, N, x3vT, N, y2, C, N, nullptr, nullptr);

        // out[b,o,n]   = relu(x[b,o,n] + W_o1 . y1^T + b_o1)   (Bt = y1, ldb=C)
        // out[b,C+o,n] = relu(x[b,o,n] + W_o2 . y2^T + b_o2)
        float* o1p = out + (size_t)b * 2 * C * N;
        float* o2p = o1p + (size_t)C * N;
        nl_gemm_wmma<MODE_FUSED><<<gO, 128, 0, stream>>>(
            wh + 3 * (size_t)wn, C, y1, C, o1p, N, C, b_o1, xb);
        nl_gemm_wmma<MODE_FUSED><<<gO, 128, 0, stream>>>(
            wh + 4 * (size_t)wn, C, y2, C, o2p, N, C, b_o2, xb);
    }
}
